// PRM_50620484550821
// MI455X (gfx1250) — compile-verified, matrix operand order checked by reading, not run on hardware
//
#include <hip/hip_runtime.h>
#include <hip/hip_bf16.h>

typedef __attribute__((ext_vector_type(16))) _Float16 v16h;
typedef __attribute__((ext_vector_type(8)))  float    v8f;
typedef __attribute__((ext_vector_type(4)))  float    v4f;

#define NP   403
#define NIN  768
#define NH1  24
#define NH2  18
#define NH3  12
#define NH4  8
#define NLAT 2
#define NB   1024
#define BT   256      // batch rows per workgroup
#define HS   32       // LDS row stride (24 cols padded to 32, pad = 0 for WMMA K-padding)

static __device__ __forceinline__ v8f wmma16(v16h a, v16h b, v8f c) {
  // D = A(16x32 f16) * B(32x16 f16) + C(16x16 f32)
  return __builtin_amdgcn_wmma_f32_16x16x32_f16(false, a, false, b, (short)0, c, false, false);
}

// Convert 8 consecutive f32 at s into f16 fragment slots [o..o+7]
static __device__ __forceinline__ void cvt8(v16h& d, int o, const float* __restrict__ s) {
  v4f q0 = *(const v4f*)s;
  v4f q1 = *(const v4f*)(s + 4);
#pragma unroll
  for (int j = 0; j < 4; ++j) {
    d[o + j]     = (_Float16)q0[j];
    d[o + 4 + j] = (_Float16)q1[j];
  }
}

template<int O, int I>
static __device__ __forceinline__ void mlayer(const float* __restrict__ w,
                                              const float* __restrict__ b,
                                              const float* in, float* out, bool relu) {
#pragma unroll
  for (int o = 0; o < O; ++o) {
    float s = b[o];
#pragma unroll
    for (int i = 0; i < I; ++i) s = fmaf(w[o * I + i], in[i], s);
    out[o] = relu ? fmaxf(s, 0.f) : s;
  }
}

__global__ __launch_bounds__(BT) void prm_autoenc(
    const float* __restrict__ x,
    const float* __restrict__ e1w, const float* __restrict__ e1b,
    const float* __restrict__ e2w, const float* __restrict__ e2b,
    const float* __restrict__ e3w, const float* __restrict__ e3b,
    const float* __restrict__ e4w, const float* __restrict__ e4b,
    const float* __restrict__ e5w, const float* __restrict__ e5b,
    const float* __restrict__ d1w, const float* __restrict__ d1b,
    const float* __restrict__ d2w, const float* __restrict__ d2b,
    const float* __restrict__ d3w, const float* __restrict__ d3b,
    const float* __restrict__ d4w, const float* __restrict__ d4b,
    const float* __restrict__ d5w, const float* __restrict__ d5b,
    float* __restrict__ recon, float* __restrict__ latent)
{
  __shared__ float hbuf[BT * HS];   // activations, rows x 32 (cols 24..31 stay 0)
  __shared__ float mw[1622];        // all middle-layer weights+biases for this p

  const int tid   = threadIdx.x;
  const int lane  = tid & 31;
  const int wave  = tid >> 5;
  const int lrow  = lane & 15;            // A: row in tile / B,C: col in tile
  const int khalf = (lane >> 4) << 3;     // 0 or 8: which K-octet this lane holds
  const int mhalf = (lane >> 4) << 3;     // C layout: lanes 16-31 hold rows M=8..15

  const int p  = blockIdx.x % NP;         // consecutive blocks share the x tile in L2
  const int tb = (blockIdx.x / NP) * BT;

  // Zero WMMA K-padding columns (24..31) of the activation buffer.
  for (int c = NH1; c < HS; ++c) hbuf[tid * HS + c] = 0.f;

  // Stage middle-layer weights into LDS (uniform-address broadcast reads later).
  {
    auto cp = [&](int off, const float* src, int n) {
      for (int i = tid; i < n; i += BT) mw[off + i] = src[i];
    };
    cp(0,    e2w + (size_t)p * NH2 * NH1, NH2 * NH1);
    cp(432,  e2b + (size_t)p * NH2,       NH2);
    cp(450,  e3w + (size_t)p * NH3 * NH2, NH3 * NH2);
    cp(666,  e3b + (size_t)p * NH3,       NH3);
    cp(678,  e4w + (size_t)p * NH4 * NH3, NH4 * NH3);
    cp(774,  e4b + (size_t)p * NH4,       NH4);
    cp(782,  e5w + (size_t)p * NLAT * NH4, NLAT * NH4);
    cp(798,  e5b + (size_t)p * NLAT,      NLAT);
    cp(800,  d1w + (size_t)p * NH4 * NLAT, NH4 * NLAT);
    cp(816,  d1b + (size_t)p * NH4,       NH4);
    cp(824,  d2w + (size_t)p * NH3 * NH4, NH3 * NH4);
    cp(920,  d2b + (size_t)p * NH3,       NH3);
    cp(932,  d3w + (size_t)p * NH2 * NH3, NH2 * NH3);
    cp(1148, d3b + (size_t)p * NH2,       NH2);
    cp(1166, d4w + (size_t)p * NH1 * NH2, NH1 * NH2);
    cp(1598, d4b + (size_t)p * NH1,       NH1);
  }

  // ---------------- Stage A: enc_fc1 [BTx768] x [768->24] via WMMA ----------------
  const int m0 = wave * 32, m1 = wave * 32 + 16;   // two 16-row M tiles per wave
  v8f acc00 = {}, acc01 = {}, acc10 = {}, acc11 = {};

  const float* xa0 = x + (size_t)(tb + m0 + lrow) * NIN + khalf;
  const float* xa1 = x + (size_t)(tb + m1 + lrow) * NIN + khalf;
  const float* wb0 = e1w + ((size_t)p * NH1 + lrow) * NIN + khalf;          // n = lrow (<16, valid)
  const int  n1   = 16 + lrow;                                              // second N tile: 16..31
  const bool n1ok = (n1 < NH1);                                             // only 16..23 valid
  const float* wb1 = e1w + ((size_t)p * NH1 + (n1ok ? n1 : 0)) * NIN + khalf;

  for (int kb = 0; kb < NIN; kb += 32) {
    v16h a0, a1, b0;
    v16h b1 = {};
    cvt8(a0, 0, xa0 + kb);  cvt8(a0, 8, xa0 + kb + 16);
    cvt8(a1, 0, xa1 + kb);  cvt8(a1, 8, xa1 + kb + 16);
    cvt8(b0, 0, wb0 + kb);  cvt8(b0, 8, wb0 + kb + 16);
    if (n1ok) { cvt8(b1, 0, wb1 + kb); cvt8(b1, 8, wb1 + kb + 16); }
    acc00 = wmma16(a0, b0, acc00);
    acc01 = wmma16(a0, b1, acc01);
    acc10 = wmma16(a1, b0, acc10);
    acc11 = wmma16(a1, b1, acc11);
  }
  {
    float bias0 = e1b[(size_t)p * NH1 + lrow];
    float bias1 = n1ok ? e1b[(size_t)p * NH1 + n1] : 0.f;
#pragma unroll
    for (int g = 0; g < 8; ++g) {
      int r0 = m0 + g + mhalf;
      int r1 = m1 + g + mhalf;
      hbuf[r0 * HS + lrow] = fmaxf(acc00[g] + bias0, 0.f);
      hbuf[r1 * HS + lrow] = fmaxf(acc10[g] + bias0, 0.f);
      if (n1ok) {
        hbuf[r0 * HS + n1] = fmaxf(acc01[g] + bias1, 0.f);
        hbuf[r1 * HS + n1] = fmaxf(acc11[g] + bias1, 0.f);
      }
    }
  }
  __syncthreads();

  // ---------------- Middle layers: one batch row per thread, weights from LDS ----------------
  {
    float va[NH1], t2[NH2], t3[NH3], t4[NH4], lt[NLAT];
#pragma unroll
    for (int i = 0; i < NH1; ++i) va[i] = hbuf[tid * HS + i];
    mlayer<NH2, NH1 >(mw + 0,    mw + 432,  va, t2, true);
    mlayer<NH3, NH2 >(mw + 450,  mw + 666,  t2, t3, true);
    mlayer<NH4, NH3 >(mw + 678,  mw + 774,  t3, t4, true);
    mlayer<NLAT, NH4>(mw + 782,  mw + 798,  t4, lt, false);   // latent, no activation
    {
      size_t bi = (size_t)(tb + tid);
      latent[bi * (NP * NLAT) + (size_t)p * NLAT + 0] = lt[0];
      latent[bi * (NP * NLAT) + (size_t)p * NLAT + 1] = lt[1];
    }
    mlayer<NH4, NLAT>(mw + 800,  mw + 816,  lt, t4, true);
    mlayer<NH3, NH4 >(mw + 824,  mw + 920,  t4, t3, true);
    mlayer<NH2, NH3 >(mw + 932,  mw + 1148, t3, t2, true);
    mlayer<NH1, NH2 >(mw + 1166, mw + 1598, t2, va, true);    // dec_fc4 output
#pragma unroll
    for (int i = 0; i < NH1; ++i) hbuf[tid * HS + i] = va[i]; // in place, row-exclusive
  }
  __syncthreads();

  // ---------------- Stage C: dec_fc5 [BTx24] x [24->768] via WMMA ----------------
  {
    v16h a0, a1;   // A fragments from LDS (K padded to 32 with zeros)
    const float* h0 = hbuf + (m0 + lrow) * HS + khalf;
    const float* h1 = hbuf + (m1 + lrow) * HS + khalf;
    cvt8(a0, 0, h0);  cvt8(a0, 8, h0 + 16);
    cvt8(a1, 0, h1);  cvt8(a1, 8, h1 + 16);

    const size_t rstride = (size_t)NP * NIN;
    for (int nt = 0; nt < NIN / 16; ++nt) {
      const int n = nt * 16 + lrow;
      const float* wp = d5w + ((size_t)p * NIN + n) * NH1 + khalf;  // w[p][n][k]
      v16h bf = {};
      cvt8(bf, 0, wp);                       // k = khalf..khalf+7   (always < 24)
      if (khalf == 0) cvt8(bf, 8, wp + 16);  // k = 16..23 valid; 24..31 stay zero
      v8f c0 = {}, c1 = {};
      c0 = wmma16(a0, bf, c0);
      c1 = wmma16(a1, bf, c1);
      const float bias = d5b[(size_t)p * NIN + n];
#pragma unroll
      for (int g = 0; g < 8; ++g) {
        int r0 = tb + m0 + g + mhalf;
        int r1 = tb + m1 + g + mhalf;
        float s0 = 1.f / (1.f + __expf(-(c0[g] + bias)));
        float s1 = 1.f / (1.f + __expf(-(c1[g] + bias)));
        __builtin_nontemporal_store(s0, &recon[(size_t)r0 * rstride + (size_t)p * NIN + n]);
        __builtin_nontemporal_store(s1, &recon[(size_t)r1 * rstride + (size_t)p * NIN + n]);
      }
    }
  }
}

extern "C" void kernel_launch(void* const* d_in, const int* in_sizes, int n_in,
                              void* d_out, int out_size, void* d_ws, size_t ws_size,
                              hipStream_t stream) {
  (void)in_sizes; (void)n_in; (void)out_size; (void)d_ws; (void)ws_size;
  const float* x   = (const float*)d_in[0];
  const float* e1w = (const float*)d_in[1];
  const float* e1b = (const float*)d_in[2];
  const float* e2w = (const float*)d_in[3];
  const float* e2b = (const float*)d_in[4];
  const float* e3w = (const float*)d_in[5];
  const float* e3b = (const float*)d_in[6];
  const float* e4w = (const float*)d_in[7];
  const float* e4b = (const float*)d_in[8];
  const float* e5w = (const float*)d_in[9];
  const float* e5b = (const float*)d_in[10];
  const float* d1w = (const float*)d_in[11];
  const float* d1b = (const float*)d_in[12];
  const float* d2w = (const float*)d_in[13];
  const float* d2b = (const float*)d_in[14];
  const float* d3w = (const float*)d_in[15];
  const float* d3b = (const float*)d_in[16];
  const float* d4w = (const float*)d_in[17];
  const float* d4b = (const float*)d_in[18];
  const float* d5w = (const float*)d_in[19];
  const float* d5b = (const float*)d_in[20];

  float* recon  = (float*)d_out;
  float* latent = recon + (size_t)NB * NP * NIN;

  dim3 grid(NP * (NB / BT));   // 403 programs x 4 batch tiles = 1612 blocks
  dim3 block(BT);              // 256 threads = 8 wave32
  prm_autoenc<<<grid, block, 0, stream>>>(
      x, e1w, e1b, e2w, e2b, e3w, e3b, e4w, e4b, e5w, e5b,
      d1w, d1b, d2w, d2b, d3w, d3b, d4w, d4b, d5w, d5b,
      recon, latent);
}
